// ActionExpertSelfBlock_40913858462205
// MI455X (gfx1250) — compile-verified, matrix operand order checked by reading, not run on hardware
//
#include <hip/hip_runtime.h>
#include <hip/hip_bf16.h>

// ---------------------------------------------------------------------------
// CDNA5 (gfx1250) transformer block: RMSNorm -> GQA attention -> RMSNorm -> SwiGLU
// All matmuls use v_wmma_f32_16x16x32_bf16 (wave32 WMMA), f32 accumulate.
// GEMM: 32x32 register tile per wave (2x2 wmma block) + global prefetch.
// Attention: flash-style, V kept in [b][hk][d][L] layout for b128 fragment loads.
// ---------------------------------------------------------------------------

typedef __attribute__((ext_vector_type(16))) __bf16 v16bf;
typedef __attribute__((ext_vector_type(8)))  float  v8f;

#define DEVI __device__ __forceinline__

union FragB {
    v16bf v;
    unsigned int u[8];
};

// float -> bf16 bits, round-to-nearest-even
DEVI unsigned short f2bf(float f) {
    unsigned int u = __float_as_uint(f);
    u += 0x7FFFu + ((u >> 16) & 1u);
    return (unsigned short)(u >> 16);
}

DEVI unsigned int pack2(unsigned short lo, unsigned short hi) {
    return (unsigned int)lo | ((unsigned int)hi << 16);
}

// K offset (relative to a 32-wide K chunk) of element pair v for this lane.
// Matches ISA 7.12.2 16-bit A-matrix 16x32 layout:
//   lanes 0-15 : VGPR0..3 -> K=0..7, VGPR4..7 -> K=16..23
//   lanes 16-31: VGPR0..3 -> K=8..15, VGPR4..7 -> K=24..31
DEVI int frag_k(int lane, int v) {
    int kb = (lane < 16) ? 0 : 8;
    return kb + ((v < 4) ? (2 * v) : (16 + 2 * (v - 4)));
}

DEVI void load_frag(FragB& f, const unsigned short* row, int lane) {
#pragma unroll
    for (int v = 0; v < 8; ++v)
        f.u[v] = *(const unsigned int*)(row + frag_k(lane, v));
}

// ---------------------------------------------------------------------------
// Weight transpose + f32 -> bf16 :  in[K][N] (row major)  ->  out[N][K]
// ---------------------------------------------------------------------------
__global__ void convT_kernel(const float* __restrict__ in,
                             unsigned short* __restrict__ out,
                             int K, int N) {
    int idx = blockIdx.x * blockDim.x + threadIdx.x;
    if (idx >= N * K) return;
    int n = idx / K;
    int k = idx - n * K;
    out[idx] = f2bf(in[(size_t)k * N + n]);
}

// V: f32 [B*L][HK][64] -> bf16 [B][HK][64][L]  (keys contiguous per (head,dim))
__global__ void convV_kernel(const float* __restrict__ vf,
                             unsigned short* __restrict__ vT) {
    const int L = 2048, HK = 5, D = 64;
    int o = blockIdx.x * blockDim.x + threadIdx.x;
    if (o >= 2 * HK * D * L) return;
    int l    = o % L;
    int rest = o / L;
    int d    = rest % D;
    int bh   = rest / D;          // b*HK + hk
    int b    = bh / HK;
    int hk   = bh % HK;
    vT[o] = f2bf(vf[((size_t)(b * L + l) * HK + hk) * D + d]);
}

// ---------------------------------------------------------------------------
// RMSNorm: one block (256 threads) per token row of 768, output bf16
// ---------------------------------------------------------------------------
__global__ void rmsnorm_kernel(const float* __restrict__ x,
                               const float* __restrict__ w,
                               unsigned short* __restrict__ out) {
    const int E = 768;
    int row = blockIdx.x;
    const float* xr = x + (size_t)row * E;
    __shared__ float red[256];
    float ss = 0.f;
    for (int i = threadIdx.x; i < E; i += 256) {
        float v = xr[i];
        ss += v * v;
    }
    red[threadIdx.x] = ss;
    __syncthreads();
    for (int st = 128; st > 0; st >>= 1) {
        if (threadIdx.x < st) red[threadIdx.x] += red[threadIdx.x + st];
        __syncthreads();
    }
    float inv = rsqrtf(red[0] / (float)E + 1.1920929e-07f);
    for (int i = threadIdx.x; i < E; i += 256)
        out[(size_t)row * E + i] = f2bf(xr[i] * inv * w[i]);
}

// ---------------------------------------------------------------------------
// WMMA GEMM: C[M][N] (f32) = A[M][K] (bf16, row major) x Bt[N][K] (bf16)
// One wave -> 32x32 output tile: 2 A-frags x 2 B-frags -> 4 wmma per K-step,
// doubling register-level reuse of each loaded fragment. Optional fused
// residual add. Prefetch next K lines via global_prefetch_b8.
// ---------------------------------------------------------------------------
__global__ void gemm_wmma_kernel(const unsigned short* __restrict__ A,
                                 const unsigned short* __restrict__ Bt,
                                 const float* __restrict__ add,
                                 float* __restrict__ C,
                                 int M, int N, int K) {
    const int WAVES = 8;
    int lane = threadIdx.x & 31;
    int w    = threadIdx.x >> 5;
    int m0 = blockIdx.y * (32 * WAVES) + w * 32;
    int n0 = blockIdx.x * 32;
    if (m0 >= M) return;
    int ln = lane & 15;

    const unsigned short* arow0 = A  + (size_t)(m0 + ln) * K;
    const unsigned short* arow1 = A  + (size_t)(m0 + 16 + ln) * K;
    const unsigned short* brow0 = Bt + (size_t)(n0 + ln) * K;
    const unsigned short* brow1 = Bt + (size_t)(n0 + 16 + ln) * K;

    v8f acc00 = {}, acc01 = {}, acc10 = {}, acc11 = {};
    for (int kk = 0; kk < K; kk += 32) {
        // prefetch 4 chunks ahead (128 bf16 = 256 B per row)
        if (kk + 128 < K) {
            __builtin_prefetch(arow0 + kk + 128, 0, 1);
            __builtin_prefetch(arow1 + kk + 128, 0, 1);
            __builtin_prefetch(brow0 + kk + 128, 0, 1);
            __builtin_prefetch(brow1 + kk + 128, 0, 1);
        }
        FragB fa0, fa1, fb0, fb1;
        load_frag(fa0, arow0 + kk, lane);
        load_frag(fa1, arow1 + kk, lane);
        load_frag(fb0, brow0 + kk, lane);
        load_frag(fb1, brow1 + kk, lane);
        acc00 = __builtin_amdgcn_wmma_f32_16x16x32_bf16(false, fa0.v, false, fb0.v, (short)0, acc00, false, false);
        acc01 = __builtin_amdgcn_wmma_f32_16x16x32_bf16(false, fa0.v, false, fb1.v, (short)0, acc01, false, false);
        acc10 = __builtin_amdgcn_wmma_f32_16x16x32_bf16(false, fa1.v, false, fb0.v, (short)0, acc10, false, false);
        acc11 = __builtin_amdgcn_wmma_f32_16x16x32_bf16(false, fa1.v, false, fb1.v, (short)0, acc11, false, false);
    }

    int hi = lane >> 4;
    v8f* accs[4] = { &acc00, &acc01, &acc10, &acc11 };
#pragma unroll
    for (int t = 0; t < 4; ++t) {
        int mrow = m0 + (t >> 1) * 16 + 8 * hi;
        int n    = n0 + (t & 1) * 16 + ln;
#pragma unroll
        for (int r = 0; r < 8; ++r) {
            size_t off = (size_t)(mrow + r) * N + n;
            float val = (*accs[t])[r];
            if (add) val += add[off];
            C[off] = val;
        }
    }
}

// ---------------------------------------------------------------------------
// RoPE + downcast: src f32 [2L][NH][64] -> dst bf16, rotation per position
// ---------------------------------------------------------------------------
__global__ void rope_kernel(const float* __restrict__ src,
                            unsigned short* __restrict__ dst, int NH) {
    const int L = 2048, D = 64, DH = 32;
    int idx = blockIdx.x * blockDim.x + threadIdx.x;
    int total = 2 * L * NH * DH;
    if (idx >= total) return;
    int d   = idx % DH;
    int tmp = idx / DH;
    int hh  = tmp % NH;
    int tok = tmp / NH;
    int pos = tok % L;
    float ts  = __powf(10000.f, (2.0f / (float)D) * (float)d);
    float rad = (float)pos / ts;
    float s = __sinf(rad), c = __cosf(rad);
    size_t base = ((size_t)tok * NH + hh) * D;
    float x1 = src[base + d];
    float x2 = src[base + DH + d];
    dst[base + d]      = f2bf(x1 * c - x2 * s);
    dst[base + DH + d] = f2bf(x2 * c + x1 * s);
}

// ---------------------------------------------------------------------------
// Flash-style causal GQA attention, one wave per (b, head, 16-query tile).
// QK^T and P*V both through v_wmma_f32_16x16x32_bf16; online softmax with
// wave32 shfl_xor row reductions; P transposed via LDS to A-fragment layout.
// V is in [b][hk][d][L] layout so P*V B-fragments load as contiguous pairs.
// ---------------------------------------------------------------------------
__global__ void attn_wmma_kernel(const unsigned short* __restrict__ qb, // [2L][15][64]
                                 const unsigned short* __restrict__ kb, // [2L][5][64]
                                 const unsigned short* __restrict__ vT, // [2][5][64][L]
                                 unsigned short* __restrict__ ctxb) {   // [2L][15][64]
    const int L = 2048, H = 15, HK = 5, D = 64;
    int lane = threadIdx.x & 31;
    int qi = blockIdx.x;       // query tile (16 rows)
    int h  = blockIdx.y;       // q head
    int b  = blockIdx.z;       // batch
    int hk = h / 3;            // GQA: 15 q heads -> 5 kv heads
    int ln = lane & 15;
    int hi = lane >> 4;

    __shared__ float lds[16 * 32];

    // Q fragments (two 32-wide chunks of head dim)
    const unsigned short* qrow = qb + ((size_t)(b * L + qi * 16 + ln) * H + h) * D;
    FragB aq0, aq1;
    load_frag(aq0, qrow, lane);
    load_frag(aq1, qrow + 32, lane);

    // V columns for this lane: four d-blocks, keys contiguous
    const unsigned short* vcol[4];
#pragma unroll
    for (int nb = 0; nb < 4; ++nb)
        vcol[nb] = vT + ((size_t)(b * HK + hk) * D + nb * 16 + ln) * L;

    v8f o[4] = {};
    float mrow[8], lrow[8];
#pragma unroll
    for (int r = 0; r < 8; ++r) { mrow[r] = -1e30f; lrow[r] = 0.f; }

    int nkb = (qi * 16 + 16 + 31) / 32;   // 32-key blocks covering causal span
    for (int j = 0; j < nkb; ++j) {
        int k0 = j * 32;

        // ---- scores: two 16x16 tiles, each over D=64 (two K=32 WMMAs) ----
        v8f s0 = {}, s1 = {};
        {
            const unsigned short* kr0 = kb + ((size_t)(b * L + k0 + ln) * HK + hk) * D;
            const unsigned short* kr1 = kb + ((size_t)(b * L + k0 + 16 + ln) * HK + hk) * D;
            FragB fb0, fb1;
            load_frag(fb0, kr0, lane);
            load_frag(fb1, kr1, lane);
            s0 = __builtin_amdgcn_wmma_f32_16x16x32_bf16(false, aq0.v, false, fb0.v, (short)0, s0, false, false);
            s1 = __builtin_amdgcn_wmma_f32_16x16x32_bf16(false, aq0.v, false, fb1.v, (short)0, s1, false, false);
            load_frag(fb0, kr0 + 32, lane);
            load_frag(fb1, kr1 + 32, lane);
            s0 = __builtin_amdgcn_wmma_f32_16x16x32_bf16(false, aq1.v, false, fb0.v, (short)0, s0, false, false);
            s1 = __builtin_amdgcn_wmma_f32_16x16x32_bf16(false, aq1.v, false, fb1.v, (short)0, s1, false, false);
        }

        // ---- scale + causal mask ----
#pragma unroll
        for (int r = 0; r < 8; ++r) {
            int qidx = qi * 16 + 8 * hi + r;
            float v0 = s0[r] * 0.125f;              // 1/sqrt(64)
            float v1 = s1[r] * 0.125f;
            if (k0 + ln      > qidx) v0 = -1e30f;
            if (k0 + 16 + ln > qidx) v1 = -1e30f;
            s0[r] = v0; s1[r] = v1;
        }

        // ---- online softmax ----
        float rm[8];
#pragma unroll
        for (int r = 0; r < 8; ++r) rm[r] = fmaxf(s0[r], s1[r]);
#pragma unroll
        for (int off = 1; off < 16; off <<= 1) {
#pragma unroll
            for (int r = 0; r < 8; ++r)
                rm[r] = fmaxf(rm[r], __shfl_xor(rm[r], off, 32));
        }
        float p0[8], p1[8], rs[8];
#pragma unroll
        for (int r = 0; r < 8; ++r) {
            float mn  = fmaxf(mrow[r], rm[r]);
            float esc = __expf(mrow[r] - mn);
            mrow[r] = mn;
            p0[r] = __expf(s0[r] - mn);
            p1[r] = __expf(s1[r] - mn);
            rs[r] = p0[r] + p1[r];
            o[0][r] *= esc; o[1][r] *= esc; o[2][r] *= esc; o[3][r] *= esc;
            lrow[r] *= esc;
        }
#pragma unroll
        for (int off = 1; off < 16; off <<= 1) {
#pragma unroll
            for (int r = 0; r < 8; ++r)
                rs[r] += __shfl_xor(rs[r], off, 32);
        }
#pragma unroll
        for (int r = 0; r < 8; ++r) lrow[r] += rs[r];

        // ---- transpose P (C layout -> A-fragment layout) through LDS ----
        __syncthreads();
#pragma unroll
        for (int r = 0; r < 8; ++r) {
            int m = 8 * hi + r;
            lds[m * 32 + ln]      = p0[r];
            lds[m * 32 + 16 + ln] = p1[r];
        }
        __syncthreads();
        FragB pa;
#pragma unroll
        for (int v = 0; v < 8; ++v) {
            int kp = frag_k(lane, v);
            pa.u[v] = pack2(f2bf(lds[ln * 32 + kp]), f2bf(lds[ln * 32 + kp + 1]));
        }

        // ---- P (16x32) * V (32x64) over four 16-column blocks ----
#pragma unroll
        for (int nb = 0; nb < 4; ++nb) {
            FragB bv;
            load_frag(bv, vcol[nb] + k0, lane);   // contiguous key pairs
            o[nb] = __builtin_amdgcn_wmma_f32_16x16x32_bf16(
                false, pa.v, false, bv.v, (short)0, o[nb], false, false);
        }
    }

    // ---- normalize + store context as bf16 (A-matrix for the O projection) ----
#pragma unroll
    for (int r = 0; r < 8; ++r) {
        float inv = 1.0f / lrow[r];
        int tok = b * L + qi * 16 + 8 * hi + r;
        unsigned short* cp = ctxb + ((size_t)tok * H + h) * D + ln;
#pragma unroll
        for (int nb = 0; nb < 4; ++nb)
            cp[nb * 16] = f2bf(o[nb][r] * inv);
    }
}

// ---------------------------------------------------------------------------
// SwiGLU activation: act = silu(g) * u, downcast to bf16
// ---------------------------------------------------------------------------
__global__ void silu_mul_kernel(const float* __restrict__ g,
                                const float* __restrict__ u,
                                unsigned short* __restrict__ out, int n) {
    int i = blockIdx.x * blockDim.x + threadIdx.x;
    if (i >= n) return;
    float gv = g[i];
    float sv = gv / (1.f + __expf(-gv));
    out[i] = f2bf(sv * u[i]);
}

// ---------------------------------------------------------------------------
// Host-side launch
// ---------------------------------------------------------------------------
extern "C" void kernel_launch(void* const* d_in, const int* in_sizes, int n_in,
                              void* d_out, int out_size, void* d_ws, size_t ws_size,
                              hipStream_t stream) {
    (void)in_sizes; (void)n_in; (void)out_size; (void)ws_size;
    const int B = 2, L = 2048, T = B * L;       // 4096 tokens
    const int E = 768, H = 15, HK = 5, D = 64;
    const int QN = H * D;                       // 960
    const int KVN = HK * D;                     // 320
    const int F = 2048;

    const float* x      = (const float*)d_in[0];
    const float* ln1_w  = (const float*)d_in[1];
    const float* wq     = (const float*)d_in[2];
    const float* wk     = (const float*)d_in[3];
    const float* wv     = (const float*)d_in[4];
    const float* wo     = (const float*)d_in[5];
    const float* ln2_w  = (const float*)d_in[6];
    const float* wg     = (const float*)d_in[7];
    const float* wu     = (const float*)d_in[8];
    const float* wd     = (const float*)d_in[9];
    float* out = (float*)d_out;

    // --- workspace bump allocator (256B aligned) ---
    char* ws = (char*)d_ws;
    size_t off = 0;
    auto alloc = [&](size_t bytes) -> void* {
        void* p = ws + off;
        off += (bytes + 255) & ~(size_t)255;
        return p;
    };
    unsigned short* wq_t = (unsigned short*)alloc((size_t)QN * E * 2);
    unsigned short* wk_t = (unsigned short*)alloc((size_t)KVN * E * 2);
    unsigned short* wv_t = (unsigned short*)alloc((size_t)KVN * E * 2);
    unsigned short* wo_t = (unsigned short*)alloc((size_t)E * QN * 2);
    unsigned short* wg_t = (unsigned short*)alloc((size_t)F * E * 2);
    unsigned short* wu_t = (unsigned short*)alloc((size_t)F * E * 2);
    unsigned short* wd_t = (unsigned short*)alloc((size_t)E * F * 2);
    unsigned short* h_bf = (unsigned short*)alloc((size_t)T * E * 2);
    float*          qf   = (float*)alloc((size_t)T * QN * 4);
    float*          kf   = (float*)alloc((size_t)T * KVN * 4);
    float*          vf   = (float*)alloc((size_t)T * KVN * 4);
    unsigned short* q_bf = (unsigned short*)alloc((size_t)T * QN * 2);
    unsigned short* k_bf = (unsigned short*)alloc((size_t)T * KVN * 2);
    unsigned short* v_T  = (unsigned short*)alloc((size_t)T * KVN * 2);
    unsigned short* ctxb = (unsigned short*)alloc((size_t)T * QN * 2);
    float*          x1   = (float*)alloc((size_t)T * E * 4);
    unsigned short* h2bf = (unsigned short*)alloc((size_t)T * E * 2);
    float*          gf   = (float*)alloc((size_t)T * F * 4);
    float*          uf   = (float*)alloc((size_t)T * F * 4);
    unsigned short* actb = (unsigned short*)alloc((size_t)T * F * 2);

    auto cdiv = [](int a, int b) { return (a + b - 1) / b; };

    // 1) weight transpose + bf16 downcast
    convT_kernel<<<cdiv(QN * E, 256), 256, 0, stream>>>(wq, wq_t, E, QN);
    convT_kernel<<<cdiv(KVN * E, 256), 256, 0, stream>>>(wk, wk_t, E, KVN);
    convT_kernel<<<cdiv(KVN * E, 256), 256, 0, stream>>>(wv, wv_t, E, KVN);
    convT_kernel<<<cdiv(E * QN, 256), 256, 0, stream>>>(wo, wo_t, QN, E);
    convT_kernel<<<cdiv(F * E, 256), 256, 0, stream>>>(wg, wg_t, E, F);
    convT_kernel<<<cdiv(F * E, 256), 256, 0, stream>>>(wu, wu_t, E, F);
    convT_kernel<<<cdiv(E * F, 256), 256, 0, stream>>>(wd, wd_t, F, E);

    // 2) RMSNorm 1
    rmsnorm_kernel<<<T, 256, 0, stream>>>(x, ln1_w, h_bf);

    // 3) QKV projections (WMMA, 32x32 tiles -> grid N/32 x M/256)
    gemm_wmma_kernel<<<dim3(QN / 32, T / 256), 256, 0, stream>>>(h_bf, wq_t, nullptr, qf, T, QN, E);
    gemm_wmma_kernel<<<dim3(KVN / 32, T / 256), 256, 0, stream>>>(h_bf, wk_t, nullptr, kf, T, KVN, E);
    gemm_wmma_kernel<<<dim3(KVN / 32, T / 256), 256, 0, stream>>>(h_bf, wv_t, nullptr, vf, T, KVN, E);

    // 4) RoPE on q,k + downcast; transpose+downcast v
    rope_kernel<<<cdiv(T * H * 32, 256), 256, 0, stream>>>(qf, q_bf, H);
    rope_kernel<<<cdiv(T * HK * 32, 256), 256, 0, stream>>>(kf, k_bf, HK);
    convV_kernel<<<cdiv(T * KVN, 256), 256, 0, stream>>>(vf, v_T);

    // 5) causal GQA flash attention (WMMA)
    attn_wmma_kernel<<<dim3(L / 16, H, B), 32, 0, stream>>>(q_bf, k_bf, v_T, ctxb);

    // 6) output projection + residual
    gemm_wmma_kernel<<<dim3(E / 32, T / 256), 256, 0, stream>>>(ctxb, wo_t, x, x1, T, E, QN);

    // 7) RMSNorm 2
    rmsnorm_kernel<<<T, 256, 0, stream>>>(x1, ln2_w, h2bf);

    // 8) SwiGLU FFN
    gemm_wmma_kernel<<<dim3(F / 32, T / 256), 256, 0, stream>>>(h2bf, wg_t, nullptr, gf, T, F, E);
    gemm_wmma_kernel<<<dim3(F / 32, T / 256), 256, 0, stream>>>(h2bf, wu_t, nullptr, uf, T, F, E);
    silu_mul_kernel<<<cdiv(T * F, 256), 256, 0, stream>>>(gf, uf, actb, T * F);
    gemm_wmma_kernel<<<dim3(E / 32, T / 256), 256, 0, stream>>>(actb, wd_t, x1, out, T, E, F);
}
